// MoE_20753281974538
// MI455X (gfx1250) — compile-verified, matrix operand order checked by reading, not run on hardware
//
#include <hip/hip_runtime.h>

// ---------------------------------------------------------------------------
// MoE: out[b,l,k] = sum_n softmax_n(x@Wg+bg) * relu(x @ We[n] + be[n])
// B*L = 8192 tokens, D_IN = D_EXP = 1024, N_EXP = 8.
// Compute-bound (137 GFLOP vs ~100MB): run the expert GEMMs on the bf16
// matrix cores (v_wmma_f32_16x16x32_bf16, f32 accumulation), everything else
// in f32.
// ---------------------------------------------------------------------------

typedef __attribute__((ext_vector_type(16))) __bf16 v16bf;
typedef __attribute__((ext_vector_type(8)))  __bf16 v8bf;
typedef __attribute__((ext_vector_type(4)))  __bf16 v4bf;
typedef __attribute__((ext_vector_type(8)))  float  v8f;
typedef __attribute__((ext_vector_type(4)))  float  v4f;

#define TOKENS 8192
#define DIN    1024
#define DEXP   1024
#define NEXP   8

// ---------------------------------------------------------------- convert x
__global__ void convert_x(const float* __restrict__ in, __bf16* __restrict__ out) {
    size_t i = ((size_t)blockIdx.x * blockDim.x + threadIdx.x) * 4;
    v4f v = *(const v4f*)(in + i);
    v4bf o = { (__bf16)v[0], (__bf16)v[1], (__bf16)v[2], (__bf16)v[3] };
    *(v4bf*)(out + i) = o;
}

// ------------------------------------------------ transpose + convert We
// We[n][d][j] (f32) -> WeT[n][j][d] (bf16), LDS-tiled so both global accesses
// are coalesced.
__global__ void transpose_we(const float* __restrict__ We, __bf16* __restrict__ WeT) {
    __shared__ float tile[32][33];
    const int n  = blockIdx.z;
    const int j0 = blockIdx.x * 32;
    const int k0 = blockIdx.y * 32;
    const int tx = threadIdx.x;      // 0..31
    const int ty = threadIdx.y;      // 0..7
    const float* src = We + ((size_t)n << 20);
#pragma unroll
    for (int i = 0; i < 32; i += 8)
        tile[ty + i][tx] = src[(size_t)(k0 + ty + i) * DEXP + j0 + tx];
    __syncthreads();
    __bf16* dst = WeT + ((size_t)n << 20);
#pragma unroll
    for (int i = 0; i < 32; i += 8)
        dst[(size_t)(j0 + ty + i) * DIN + k0 + tx] = (__bf16)tile[tx][ty + i];
}

// ------------------------------------------------------------- gating softmax
// One wave32 per token: full-precision f32 dot products + softmax over 8.
__global__ void gate_kernel(const float* __restrict__ x, const float* __restrict__ Wg,
                            const float* __restrict__ bg, float* __restrict__ gate) {
    const int token = blockIdx.x * (blockDim.x >> 5) + (threadIdx.x >> 5);
    const int lane  = threadIdx.x & 31;
    float acc[NEXP];
#pragma unroll
    for (int n = 0; n < NEXP; ++n) acc[n] = 0.0f;
    const float* xr = x + (size_t)token * DIN;
    for (int d = lane; d < DIN; d += 32) {
        const float xv = xr[d];
        const float* wr = Wg + (size_t)d * NEXP;
#pragma unroll
        for (int n = 0; n < NEXP; ++n) acc[n] = fmaf(xv, wr[n], acc[n]);
    }
#pragma unroll
    for (int n = 0; n < NEXP; ++n)
        for (int off = 16; off > 0; off >>= 1)
            acc[n] += __shfl_xor(acc[n], off, 32);
    if (lane == 0) {
        float mx = -3.4e38f;
#pragma unroll
        for (int n = 0; n < NEXP; ++n) { acc[n] += bg[n]; mx = fmaxf(mx, acc[n]); }
        float e[NEXP], s = 0.0f;
#pragma unroll
        for (int n = 0; n < NEXP; ++n) { e[n] = __expf(acc[n] - mx); s += e[n]; }
        const float inv = 1.0f / s;
#pragma unroll
        for (int n = 0; n < NEXP; ++n) gate[(size_t)token * NEXP + n] = e[n] * inv;
    }
}

// ------------------------------------------------------------- main WMMA GEMM
// Block: 256 threads = 8 waves in a 4(M) x 2(N) grid. WG tile 128x128,
// wave tile 32x64 = 2x4 WMMA 16x16 tiles. Experts outer, K=1024 inner,
// bias+relu+gate epilogue per expert into persistent f32 accumulators.
//
// bf16 16x32 A/B fragment layout (ISA 7.12.2): lane<16 holds row/col (lane&15)
// with K = k..k+7 (bytes 0..15) and K = k+16..k+23 (bytes 32..47); lane>=16
// holds K = k+8..k+15 and k+24..k+31  -> two contiguous 16B loads per lane.
__global__ __launch_bounds__(256, 1) void moe_wmma(
    const __bf16* __restrict__ xb, const __bf16* __restrict__ WeT,
    const float* __restrict__ be, const float* __restrict__ gate,
    float* __restrict__ out) {
    const int lane  = threadIdx.x & 31;
    const int wave  = threadIdx.x >> 5;
    const int waveM = wave >> 1;            // 0..3
    const int waveN = wave & 1;             // 0..1
    const int lr    = lane & 15;
    const int lh    = lane >> 4;            // 0 or 1

    const int mBase = blockIdx.x * 128 + waveM * 32;  // token rows
    const int jBase = blockIdx.y * 128 + waveN * 64;  // output cols

    const v8f zero = {};
    v8f acc[2][4];
#pragma unroll
    for (int t = 0; t < 2; ++t)
#pragma unroll
        for (int u = 0; u < 4; ++u) acc[t][u] = zero;

    const __bf16* ap[2];
#pragma unroll
    for (int t = 0; t < 2; ++t)
        ap[t] = xb + (size_t)(mBase + t * 16 + lr) * DIN + lh * 8;

#pragma unroll 1
    for (int n = 0; n < NEXP; ++n) {
        const __bf16* wb = WeT + ((size_t)n << 20) + lh * 8;
        const __bf16* bp[4];
#pragma unroll
        for (int u = 0; u < 4; ++u)
            bp[u] = wb + (size_t)(jBase + u * 16 + lr) * DIN;

        v8f h[2][4];
#pragma unroll
        for (int t = 0; t < 2; ++t)
#pragma unroll
            for (int u = 0; u < 4; ++u) h[t][u] = zero;

#pragma unroll 2
        for (int k = 0; k < DIN; k += 32) {
            v16bf a[2], b[4];
#pragma unroll
            for (int t = 0; t < 2; ++t) {
                v8bf lo = *(const v8bf*)(ap[t] + k);
                v8bf hi = *(const v8bf*)(ap[t] + k + 16);
                a[t] = __builtin_shufflevector(lo, hi, 0, 1, 2, 3, 4, 5, 6, 7,
                                               8, 9, 10, 11, 12, 13, 14, 15);
            }
#pragma unroll
            for (int u = 0; u < 4; ++u) {
                v8bf lo = *(const v8bf*)(bp[u] + k);
                v8bf hi = *(const v8bf*)(bp[u] + k + 16);
                b[u] = __builtin_shufflevector(lo, hi, 0, 1, 2, 3, 4, 5, 6, 7,
                                               8, 9, 10, 11, 12, 13, 14, 15);
            }
#pragma unroll
            for (int t = 0; t < 2; ++t)
#pragma unroll
                for (int u = 0; u < 4; ++u)
                    h[t][u] = __builtin_amdgcn_wmma_f32_16x16x32_bf16(
                        false, a[t], false, b[u], (short)0, h[t][u], false, false);
        }

        // epilogue: bias -> relu -> gate-weighted accumulate
        // C/D layout: value v of this lane is row (lh*8 + v), col lr of the tile.
#pragma unroll
        for (int t = 0; t < 2; ++t) {
            float g[8];
#pragma unroll
            for (int v = 0; v < 8; ++v)
                g[v] = gate[(size_t)(mBase + t * 16 + lh * 8 + v) * NEXP + n];
#pragma unroll
            for (int u = 0; u < 4; ++u) {
                const float bias = be[n * DEXP + jBase + u * 16 + lr];
#pragma unroll
                for (int v = 0; v < 8; ++v)
                    acc[t][u][v] += g[v] * fmaxf(h[t][u][v] + bias, 0.0f);
            }
        }
    }

#pragma unroll
    for (int t = 0; t < 2; ++t)
#pragma unroll
        for (int u = 0; u < 4; ++u)
#pragma unroll
            for (int v = 0; v < 8; ++v)
                out[(size_t)(mBase + t * 16 + lh * 8 + v) * DEXP + jBase + u * 16 + lr] =
                    acc[t][u][v];
}

// ---------------------------------------------------------------------------
extern "C" void kernel_launch(void* const* d_in, const int* in_sizes, int n_in,
                              void* d_out, int out_size, void* d_ws, size_t ws_size,
                              hipStream_t stream) {
    const float* x  = (const float*)d_in[0];  // (4,2048,1024)
    const float* We = (const float*)d_in[1];  // (8,1024,1024)
    const float* be = (const float*)d_in[2];  // (8,1024)
    const float* Wg = (const float*)d_in[3];  // (1024,8)
    const float* bg = (const float*)d_in[4];  // (8)
    float* out = (float*)d_out;               // (4,2048,1024)

    char* ws = (char*)d_ws;
    __bf16* xb   = (__bf16*)(ws);                          // 16 MB
    __bf16* WeT  = (__bf16*)(ws + ((size_t)16 << 20));     // 16 MB
    float*  gate = (float*)(ws + ((size_t)32 << 20));      // 256 KB

    // 1) x -> bf16
    convert_x<<<(TOKENS * DIN) / (256 * 4), 256, 0, stream>>>(x, xb);
    // 2) We -> bf16, transposed to [n][j][k]
    transpose_we<<<dim3(DEXP / 32, DIN / 32, NEXP), dim3(32, 8), 0, stream>>>(We, WeT);
    // 3) gating softmax (f32)
    gate_kernel<<<TOKENS / 8, 256, 0, stream>>>(x, Wg, bg, gate);
    // 4) fused expert-GEMM + bias + relu + gated sum (bf16 WMMA, f32 accum)
    moe_wmma<<<dim3(TOKENS / 128, DEXP / 128), 256, 0, stream>>>(xb, WeT, be, gate, out);
}
